// FSMNMultiHeadAttention_927712936519
// MI455X (gfx1250) — compile-verified, matrix operand order checked by reading, not run on hardware
//
#include <hip/hip_runtime.h>
#include <hip/hip_bf16.h>

// Problem constants (match reference)
#define Bb 4
#define Tt 1024
#define Dd 1024
#define Hh 16
#define Kk 31
#define DhC 64

typedef __attribute__((ext_vector_type(16))) __bf16 v16bf;
typedef __attribute__((ext_vector_type(8)))  __bf16 v8bf;
typedef __attribute__((ext_vector_type(8)))  float  v8f;

// float -> bf16 round-to-nearest-even (bit-level; only used in one-shot conversion
// passes and the softmax P staging, never in GEMM inner loops).
static __device__ __forceinline__ __bf16 f2bf(float f) {
  union { float f; unsigned u; } x; x.f = f;
  unsigned r = (x.u + 0x7FFFu + ((x.u >> 16) & 1u)) >> 16;
  unsigned short s = (unsigned short)r;
  __bf16 out;
  __builtin_memcpy(&out, &s, sizeof(out));
  return out;
}

static __device__ __forceinline__ v8f wmma_bf16(v16bf a, v16bf b, v8f c) {
  return __builtin_amdgcn_wmma_f32_16x16x32_bf16(false, a, false, b, (short)0, c, false, false);
}

// Load a 16x32 WMMA A/B fragment straight from row-major bf16 memory.
// CDNA5 16-bit fragment layout (wave32): lane = hf*16 + rr holds row rr,
// elements j<8 -> k = k0 + hf*8 + j ; j>=8 -> k = k0 + 16 + hf*8 + (j-8).
// Each 8-element run is 16 contiguous bytes -> two global_load_b128, no VALU.
static __device__ __forceinline__ v16bf load_frag_bf(const __bf16* __restrict__ base,
                                                     int ld, int row, int k0, int hf) {
  const __bf16* p = base + (size_t)row * ld + k0 + hf * 8;
  v8bf lo = *(const v8bf*)(p);
  v8bf hi = *(const v8bf*)(p + 16);
  v16bf f;
#pragma unroll
  for (int j = 0; j < 8; ++j) f[j] = lo[j];
#pragma unroll
  for (int j = 0; j < 8; ++j) f[8 + j] = hi[j];
  return f;
}

// C = A(MxK,bf16) * W(NxK,bf16)^T + bias[n] (+ addend[m,n]); one wave -> 32x64 tile.
// 8 independent WMMA chains per k-step: 2 A-fragments x 4 B-fragments.
__global__ void __launch_bounds__(32)
gemm_bf_kernel(const __bf16* __restrict__ A, const __bf16* __restrict__ W,
               const float* __restrict__ bias, const float* __restrict__ addend,
               float* __restrict__ C, int M, int N, int Kd) {
  const int lane = threadIdx.x;
  const int hf = lane >> 4, rr = lane & 15;
  const int n0 = blockIdx.x * 64;
  const int m0 = blockIdx.y * 32;
  v8f acc0_0 = {}, acc0_1 = {}, acc0_2 = {}, acc0_3 = {};
  v8f acc1_0 = {}, acc1_1 = {}, acc1_2 = {}, acc1_3 = {};
  for (int k0 = 0; k0 < Kd; k0 += 32) {
    // Speculative prefetch of the next k-panel (safe past end; silently dropped).
    __builtin_prefetch(A + (size_t)(m0 + rr) * Kd + k0 + 32, 0, 1);
    __builtin_prefetch(W + (size_t)(n0 + rr) * Kd + k0 + 32, 0, 1);
    v16bf a0 = load_frag_bf(A, Kd, m0 +      rr, k0, hf);
    v16bf a1 = load_frag_bf(A, Kd, m0 + 16 + rr, k0, hf);
    v16bf b0 = load_frag_bf(W, Kd, n0 +      rr, k0, hf);
    v16bf b1 = load_frag_bf(W, Kd, n0 + 16 + rr, k0, hf);
    v16bf b2 = load_frag_bf(W, Kd, n0 + 32 + rr, k0, hf);
    v16bf b3 = load_frag_bf(W, Kd, n0 + 48 + rr, k0, hf);
    acc0_0 = wmma_bf16(a0, b0, acc0_0);
    acc1_0 = wmma_bf16(a1, b0, acc1_0);
    acc0_1 = wmma_bf16(a0, b1, acc0_1);
    acc1_1 = wmma_bf16(a1, b1, acc1_1);
    acc0_2 = wmma_bf16(a0, b2, acc0_2);
    acc1_2 = wmma_bf16(a1, b2, acc1_2);
    acc0_3 = wmma_bf16(a0, b3, acc0_3);
    acc1_3 = wmma_bf16(a1, b3, acc1_3);
  }
#pragma unroll
  for (int mt = 0; mt < 2; ++mt) {
#pragma unroll
    for (int t = 0; t < 4; ++t) {
      v8f acc;
      if (mt == 0) acc = (t == 0) ? acc0_0 : (t == 1) ? acc0_1 : (t == 2) ? acc0_2 : acc0_3;
      else         acc = (t == 0) ? acc1_0 : (t == 1) ? acc1_1 : (t == 2) ? acc1_2 : acc1_3;
      int n = n0 + t * 16 + rr;
      float bn = bias ? bias[n] : 0.0f;
#pragma unroll
      for (int vv = 0; vv < 8; ++vv) {
        int m = m0 + mt * 16 + hf * 8 + vv;   // C layout: M = vgpr + 8*half
        size_t off = (size_t)m * N + n;
        float val = acc[vv] + bn;
        if (addend) val += addend[off];
        C[off] = val;
      }
    }
  }
}

// One-shot f32 -> bf16 conversion (x and the four weight matrices).
__global__ void cvt_bf16_kernel(const float* __restrict__ in, __bf16* __restrict__ out,
                                size_t n) {
  size_t i = ((size_t)blockIdx.x * blockDim.x + threadIdx.x) * 2;
  if (i + 1 < n) { out[i] = f2bf(in[i]); out[i + 1] = f2bf(in[i + 1]); }
  else if (i < n) out[i] = f2bf(in[i]);
}

// RoPE on f32 q/k, emitting bf16 copies for the WMMA attention kernel.
__global__ void rope_cvt_kernel(const float* __restrict__ q, const float* __restrict__ k,
                                const float* __restrict__ fcos, const float* __restrict__ fsin,
                                __bf16* __restrict__ qbf, __bf16* __restrict__ kbf) {
  size_t idx = (size_t)blockIdx.x * blockDim.x + threadIdx.x;
  const size_t total = (size_t)Bb * Tt * Hh * (DhC / 2);
  if (idx >= total) return;
  int i = (int)(idx & 31);
  int h = (int)((idx >> 5) & (Hh - 1));
  size_t bt = idx >> 9;                 // / (32*H)
  int t = (int)(bt & (Tt - 1));
  size_t off = (bt * Hh + h) * DhC;
  float c0 = fcos[t * DhC + i],      s0 = fsin[t * DhC + i];
  float c1 = fcos[t * DhC + i + 32], s1 = fsin[t * DhC + i + 32];
  float q0 = q[off + i], q1 = q[off + i + 32];
  qbf[off + i]      = f2bf(q0 * c0 - q1 * s0);
  qbf[off + i + 32] = f2bf(q1 * c1 + q0 * s1);
  float k0v = k[off + i], k1v = k[off + i + 32];
  kbf[off + i]      = f2bf(k0v * c0 - k1v * s0);
  kbf[off + i + 32] = f2bf(k1v * c1 + k0v * s1);
}

// fsm = (depthwise_conv_K31(v*mask_pad) + v*mask_pad) * mask_pad ; v: (B,T,D), f32.
__global__ void fsmn_kernel(const float* __restrict__ v, const float* __restrict__ w,
                            const float* __restrict__ mask_pad, float* __restrict__ fsm) {
  size_t idx = (size_t)blockIdx.x * blockDim.x + threadIdx.x;
  const size_t total = (size_t)Bb * Tt * Dd;
  if (idx >= total) return;
  int d = (int)(idx & (Dd - 1));
  int t = (int)((idx >> 10) & (Tt - 1));
  int b = (int)(idx >> 20);
  float acc = 0.0f;
#pragma unroll
  for (int j = 0; j < Kk; ++j) {
    int tt = t + j - 15;                 // LPAD = 15
    if (tt >= 0 && tt < Tt)
      acc += v[((size_t)(b * Tt + tt) << 10) + d] * mask_pad[b * Tt + tt] * w[d * Kk + j];
  }
  float mp = mask_pad[b * Tt + t];
  float vf = v[idx] * mp;
  fsm[idx] = (acc + vf) * mp;
}

// v (B,T,H,Dh) f32 -> vT (B, H*Dh, T) bf16, so P@V B-fragments are contiguous.
__global__ void vtrans_kernel(const float* __restrict__ v, __bf16* __restrict__ vT) {
  size_t idx = (size_t)blockIdx.x * blockDim.x + threadIdx.x;
  const size_t total = (size_t)Bb * Tt * Dd;
  if (idx >= total) return;
  int d = (int)(idx & (Dd - 1));
  int t = (int)((idx >> 10) & (Tt - 1));
  int b = (int)(idx >> 20);
  vT[((size_t)b * Dd + d) * Tt + t] = f2bf(v[idx]);
}

// Fused flash attention: writes masked pre-softmax logits (qk output, f32) AND
// computes softmax(qk) @ V with online normalization; wv emitted in bf16.
// One wave = 16 q rows for one (b,h).
__global__ void __launch_bounds__(32)
attn_kernel(const __bf16* __restrict__ q, const __bf16* __restrict__ k,
            const __bf16* __restrict__ vT, float* __restrict__ qk_out,
            __bf16* __restrict__ wv_out) {
  __shared__ __bf16 Plds[16][32];
  const int lane = threadIdx.x;
  const int hf = lane >> 4, rr = lane & 15;
  const int qt = blockIdx.x, h = blockIdx.y, b = blockIdx.z;

  const __bf16* qbase = q + ((size_t)(b * Tt + qt * 16) * Hh + h) * DhC;
  v16bf qa0 = load_frag_bf(qbase, Dd, rr, 0, hf);    // dh 0..31
  v16bf qa1 = load_frag_bf(qbase, Dd, rr, 32, hf);   // dh 32..63

  v8f acc0 = {}, acc1 = {}, acc2 = {}, acc3 = {};    // 16x64 O accumulator
  float mrow[8], lrow[8];
#pragma unroll
  for (int i = 0; i < 8; ++i) { mrow[i] = -3.0e38f; lrow[i] = 0.0f; }

  float* qkrow = qk_out + (size_t)(b * Hh + h) * Tt * Tt;
  const __bf16* vtb = vT + ((size_t)b * Dd + h * DhC) * Tt;  // rows = dh, ld = T

  for (int c = 0; c < Tt; c += 32) {
    // ---- S = (q kT)/8 + causal mask, two 16x16 tiles (cols c..c+31) ----
    const __bf16* kb0p = k + ((size_t)(b * Tt + c) * Hh + h) * DhC;
    const __bf16* kb1p = k + ((size_t)(b * Tt + c + 16) * Hh + h) * DhC;
    v16bf kb00 = load_frag_bf(kb0p, Dd, rr, 0, hf);
    v16bf kb01 = load_frag_bf(kb0p, Dd, rr, 32, hf);
    v16bf kb10 = load_frag_bf(kb1p, Dd, rr, 0, hf);
    v16bf kb11 = load_frag_bf(kb1p, Dd, rr, 32, hf);
    v8f s0 = {}; s0 = wmma_bf16(qa0, kb00, s0); s0 = wmma_bf16(qa1, kb01, s0);
    v8f s1 = {}; s1 = wmma_bf16(qa0, kb10, s1); s1 = wmma_bf16(qa1, kb11, s1);

    // ---- mask + write qk, online softmax, stage P tile to LDS ----
#pragma unroll
    for (int vv = 0; vv < 8; ++vv) {
      int trow = qt * 16 + hf * 8 + vv;
      int c0 = c + rr, c1 = c + 16 + rr;
      float p0 = s0[vv] * 0.125f + (c0 <= trow ? 0.0f : -1.0e9f);
      float p1 = s1[vv] * 0.125f + (c1 <= trow ? 0.0f : -1.0e9f);
      qkrow[(size_t)trow * Tt + c0] = p0;
      qkrow[(size_t)trow * Tt + c1] = p1;

      float tmax = fmaxf(p0, p1);
#pragma unroll
      for (int off = 8; off >= 1; off >>= 1)
        tmax = fmaxf(tmax, __shfl_xor(tmax, off, 16));   // row lives in a 16-lane half
      float mnew = fmaxf(mrow[vv], tmax);
      float e0 = __expf(p0 - mnew), e1 = __expf(p1 - mnew);
      float rs = e0 + e1;
#pragma unroll
      for (int off = 8; off >= 1; off >>= 1)
        rs += __shfl_xor(rs, off, 16);
      float corr = __expf(mrow[vv] - mnew);
      lrow[vv] = lrow[vv] * corr + rs;
      mrow[vv] = mnew;
      acc0[vv] *= corr; acc1[vv] *= corr; acc2[vv] *= corr; acc3[vv] *= corr;

      int row = hf * 8 + vv;
      Plds[row][rr]      = f2bf(e0);
      Plds[row][16 + rr] = f2bf(e1);
    }
    asm volatile("s_wait_dscnt 0x0" ::: "memory");   // P staged (same-wave LDS in-order)

    // ---- P (16x32) from LDS as A fragment ----
    v16bf pa;
#pragma unroll
    for (int j = 0; j < 8; ++j) pa[j]     = Plds[rr][hf * 8 + j];
#pragma unroll
    for (int j = 0; j < 8; ++j) pa[8 + j] = Plds[rr][16 + hf * 8 + j];

    // ---- O += P @ V[c..c+31, :] : vT rows are dh, contiguous in t ----
    acc0 = wmma_bf16(pa, load_frag_bf(vtb + (size_t)( 0) * Tt + c, Tt, rr, 0, hf), acc0);
    acc1 = wmma_bf16(pa, load_frag_bf(vtb + (size_t)(16) * Tt + c, Tt, rr, 0, hf), acc1);
    acc2 = wmma_bf16(pa, load_frag_bf(vtb + (size_t)(32) * Tt + c, Tt, rr, 0, hf), acc2);
    acc3 = wmma_bf16(pa, load_frag_bf(vtb + (size_t)(48) * Tt + c, Tt, rr, 0, hf), acc3);
  }

  // ---- normalize and write wv (B,T,H,Dh) in bf16 for the output GEMM ----
  __bf16* wbase = wv_out + ((size_t)(b * Tt + qt * 16) * Hh + h) * DhC;
#pragma unroll
  for (int vv = 0; vv < 8; ++vv) {
    int rowl = hf * 8 + vv;
    float inv = 1.0f / lrow[vv];
    wbase[(size_t)rowl * Dd + rr + 0]  = f2bf(acc0[vv] * inv);
    wbase[(size_t)rowl * Dd + rr + 16] = f2bf(acc1[vv] * inv);
    wbase[(size_t)rowl * Dd + rr + 32] = f2bf(acc2[vv] * inv);
    wbase[(size_t)rowl * Dd + rr + 48] = f2bf(acc3[vv] * inv);
  }
}

extern "C" void kernel_launch(void* const* d_in, const int* in_sizes, int n_in,
                              void* d_out, int out_size, void* d_ws, size_t ws_size,
                              hipStream_t stream) {
  const float* x        = (const float*)d_in[0];
  // d_in[1] = mask (causal) — recomputed inline
  const float* mask_pad = (const float*)d_in[2];
  const float* fcos     = (const float*)d_in[3];
  const float* fsin     = (const float*)d_in[4];
  const float* Wq       = (const float*)d_in[5];
  const float* bq       = (const float*)d_in[6];
  const float* Wk       = (const float*)d_in[7];
  const float* Wv       = (const float*)d_in[8];
  const float* bv       = (const float*)d_in[9];
  const float* Wo       = (const float*)d_in[10];
  const float* bo       = (const float*)d_in[11];
  const float* fsmnw    = (const float*)d_in[12];

  float* out = (float*)d_out;                      // (B,T,D)
  float* qk  = out + (size_t)Bb * Tt * Dd;         // (B,H,T,T)

  const size_t MBy = (size_t)1 << 20;
  char* wsb = (char*)d_ws;
  // f32 scratch
  float* qf   = (float*)(wsb + 0 * MBy);    // 16 MB
  float* kf   = (float*)(wsb + 16 * MBy);   // 16 MB
  float* vf   = (float*)(wsb + 32 * MBy);   // 16 MB
  float* fsm  = (float*)(wsb + 48 * MBy);   // 16 MB
  // bf16 scratch
  __bf16* xbf  = (__bf16*)(wsb + 64 * MBy); //  8 MB
  __bf16* wqb  = (__bf16*)(wsb + 72 * MBy); //  2 MB each
  __bf16* wkb  = (__bf16*)(wsb + 74 * MBy);
  __bf16* wvb  = (__bf16*)(wsb + 76 * MBy);
  __bf16* wob  = (__bf16*)(wsb + 78 * MBy);
  __bf16* qbf  = (__bf16*)(wsb + 80 * MBy); //  8 MB
  __bf16* kbf  = (__bf16*)(wsb + 88 * MBy); //  8 MB
  __bf16* vTbf = (__bf16*)(wsb + 96 * MBy); //  8 MB
  __bf16* wvbf = (__bf16*)(wsb + 104 * MBy);//  8 MB (112 MB total)

  const size_t nX = (size_t)Bb * Tt * Dd;          // 4M
  const size_t nW = (size_t)Dd * Dd;               // 1M

  // One-shot conversions (each source element converted exactly once)
  cvt_bf16_kernel<<<(unsigned)((nX / 2 + 255) / 256), 256, 0, stream>>>(x, xbf, nX);
  cvt_bf16_kernel<<<(unsigned)((nW / 2 + 255) / 256), 256, 0, stream>>>(Wq, wqb, nW);
  cvt_bf16_kernel<<<(unsigned)((nW / 2 + 255) / 256), 256, 0, stream>>>(Wk, wkb, nW);
  cvt_bf16_kernel<<<(unsigned)((nW / 2 + 255) / 256), 256, 0, stream>>>(Wv, wvb, nW);
  cvt_bf16_kernel<<<(unsigned)((nW / 2 + 255) / 256), 256, 0, stream>>>(Wo, wob, nW);

  dim3 gGemm(Dd / 64, (Bb * Tt) / 32);
  gemm_bf_kernel<<<gGemm, 32, 0, stream>>>(xbf, wqb, bq, nullptr, qf, Bb * Tt, Dd, Dd);
  gemm_bf_kernel<<<gGemm, 32, 0, stream>>>(xbf, wkb, nullptr, nullptr, kf, Bb * Tt, Dd, Dd);
  gemm_bf_kernel<<<gGemm, 32, 0, stream>>>(xbf, wvb, bv, nullptr, vf, Bb * Tt, Dd, Dd);

  size_t nrope = (size_t)Bb * Tt * Hh * (DhC / 2);
  rope_cvt_kernel<<<(unsigned)((nrope + 255) / 256), 256, 0, stream>>>(qf, kf, fcos, fsin,
                                                                      qbf, kbf);
  fsmn_kernel<<<(unsigned)((nX + 255) / 256), 256, 0, stream>>>(vf, fsmnw, mask_pad, fsm);
  vtrans_kernel<<<(unsigned)((nX + 255) / 256), 256, 0, stream>>>(vf, vTbf);

  dim3 gAttn(Tt / 16, Hh, Bb);
  attn_kernel<<<gAttn, 32, 0, stream>>>(qbf, kbf, vTbf, qk, wvbf);

  gemm_bf_kernel<<<gGemm, 32, 0, stream>>>(wvbf, wob, bo, fsm, out, Bb * Tt, Dd, Dd);
}
